// DiTFlowModel_44315472560570
// MI455X (gfx1250) — matrix operand order repaired and checked by reading, not compiled, not run on hardware
//
#include <hip/hip_runtime.h>
#include <hip/hip_bf16.h>
#include <math.h>

// ---------------------------------------------------------------------------
// Types for CDNA5 WMMA (wave32, 16x16x32 bf16 -> f32)
// ---------------------------------------------------------------------------
typedef __bf16 bf16_t;
typedef __attribute__((ext_vector_type(16))) __bf16 bf16x16;
typedef __attribute__((ext_vector_type(8)))  __bf16 bf16x8;
typedef __attribute__((ext_vector_type(4)))  __bf16 bf16x4;
typedef __attribute__((ext_vector_type(8)))  float  floatx8;

// exact parameter vector type of __builtin_amdgcn_global_load_async_to_lds_b128
typedef __attribute__((__vector_size__(4 * sizeof(int)))) int ai4;

#define BM 128
#define BN 128
#define BK 32
#define LDT 40   // padded LDS row stride (bf16 elems): 80B rows -> conflict-friendly

// gfx1250 async global->LDS copy (ASYNCcnt path), with portable fallback
#if defined(__has_builtin)
#  if __has_builtin(__builtin_amdgcn_global_load_async_to_lds_b128)
#    define HAVE_ASYNC_LDS 1
#  else
#    define HAVE_ASYNC_LDS 0
#  endif
#else
#  define HAVE_ASYNC_LDS 0
#endif

__device__ __forceinline__ void async_copy_b128(const bf16_t* g, bf16_t* l) {
#if HAVE_ASYNC_LDS
    __builtin_amdgcn_global_load_async_to_lds_b128(
        (__attribute__((address_space(1))) ai4*)g,
        (__attribute__((address_space(3))) ai4*)l, 0, 0);
#else
    *(bf16x8*)l = *(const bf16x8*)g;
#endif
}

__device__ __forceinline__ void wait_async0() {
#if HAVE_ASYNC_LDS
#  if __has_builtin(__builtin_amdgcn_s_wait_asynccnt)
    __builtin_amdgcn_s_wait_asynccnt(0);
#  else
    asm volatile("s_wait_asynccnt 0x0" ::: "memory");
#  endif
#endif
}

__device__ __forceinline__ float gelu_exact(float x) {
    return 0.5f * x * (1.0f + erff(x * 0.70710678118654752440f));
}

// ---------------------------------------------------------------------------
// fp32 -> bf16 weight conversion (one pass; weights stay L2-resident)
// ---------------------------------------------------------------------------
__global__ __launch_bounds__(256) void f32_to_bf16_kernel(
    const float* __restrict__ src, bf16_t* __restrict__ dst, int n4)
{
    const int i = blockIdx.x * 256 + threadIdx.x;
    if (i >= n4) return;
    const float4 v = ((const float4*)src)[i];
    bf16x4 o = { (bf16_t)v.x, (bf16_t)v.y, (bf16_t)v.z, (bf16_t)v.w };
    ((bf16x4*)dst)[i] = o;
}

// ---------------------------------------------------------------------------
// C[M,N] = act( A[M,K] @ W[N,K]^T + bias[N] )
// A: fp32 (converted to bf16 while staging), W: bf16 (async-copied to LDS).
// Requires: M % 128 == 0, N % 128 == 0, K % 32 == 0. act: 0=none, 1=GELU.
// Double-buffered LDS: stage tile t+1 while WMMAs consume tile t.
// ---------------------------------------------------------------------------
__global__ __launch_bounds__(256) void gemm_bias_act_kernel(
    const float* __restrict__ A, const bf16_t* __restrict__ W,
    const float* __restrict__ bias, float* __restrict__ C,
    int M, int N, int K, int ldc, int act)
{
    __shared__ bf16_t As[2][BM * LDT];
    __shared__ bf16_t Ws[2][BN * LDT];

    const int tid  = threadIdx.x;
    const int lane = tid & 31;
    const int wave = tid >> 5;    // 8 waves
    const int wm   = wave & 3;    // 4 waves along M: 32 rows each
    const int wn   = wave >> 2;   // 2 waves along N: 64 cols each
    const int bm   = blockIdx.y * BM;
    const int bn   = blockIdx.x * BN;

    floatx8 acc[2][4];
    {
        const floatx8 z = {0.f,0.f,0.f,0.f,0.f,0.f,0.f,0.f};
#pragma unroll
        for (int mi = 0; mi < 2; ++mi)
#pragma unroll
            for (int ni = 0; ni < 4; ++ni) acc[mi][ni] = z;
    }

    const int rsel = lane & 15;          // row (A) / col (B) within 16-tile
    const int kh8  = (lane >> 4) * 8;    // A-frag k base (ISA 16-bit A layout)
    const int kh16 = (lane >> 4) * 16;   // unused: B frags also use kh8 (same k order as A)

    // stage one 128x32 k-tile pair into LDS buffer `b`
    auto stage = [&](int k0, int b) {
        // W tile: bf16, 512 x 16B chunks -> async copy straight into LDS
#pragma unroll
        for (int i = 0; i < 2; ++i) {
            const int c   = tid + (i << 8);
            const int row = c >> 2;
            const int c8  = (c & 3) << 3;
            async_copy_b128(W + (size_t)(bn + row) * K + k0 + c8,
                            &Ws[b][row * LDT + c8]);
        }
        // A tile: fp32 -> bf16 convert while staging (1024 float4 chunks)
#pragma unroll
        for (int i = 0; i < 4; ++i) {
            const int idx = tid + (i << 8);
            const int row = idx >> 3;
            const int c4  = (idx & 7) << 2;
            const float* gp = A + (size_t)(bm + row) * K + k0 + c4;
            const float4 av = *(const float4*)gp;
            bf16x4 ap = { (bf16_t)av.x, (bf16_t)av.y, (bf16_t)av.z, (bf16_t)av.w };
            *(bf16x4*)(&As[b][row * LDT + c4]) = ap;
            if (k0 + BK < K)  // near-scope prefetch of the following A k-tile
                __builtin_prefetch(gp + BK, 0, 3);
        }
    };

    int buf = 0;
    stage(0, 0);
    wait_async0();
    __syncthreads();

    for (int k0 = 0; k0 < K; k0 += BK) {
        const bool more = (k0 + BK) < K;
        if (more) stage(k0 + BK, buf ^ 1);

        // ---- A fragments: lane holds row m=rsel; elems 0..7 = K kh8..kh8+7,
        //      elems 8..15 = K 16+kh8..16+kh8+7
        bf16x16 afrag[2];
#pragma unroll
        for (int mi = 0; mi < 2; ++mi) {
            const bf16_t* base = &As[buf][(wm * 32 + mi * 16 + rsel) * LDT + kh8];
            bf16x8 lo = *(const bf16x8*)(base);
            bf16x8 hi = *(const bf16x8*)(base + 16);
            afrag[mi] = __builtin_shufflevector(lo, hi,
                0,1,2,3,4,5,6,7,8,9,10,11,12,13,14,15);
        }
        // ---- B fragments: lane holds col n=rsel (row of W); same k order as A
        bf16x16 wfrag[4];
#pragma unroll
        for (int ni = 0; ni < 4; ++ni) {
            const bf16_t* base = &Ws[buf][(wn * 64 + ni * 16 + rsel) * LDT + kh8];
            bf16x8 lo = *(const bf16x8*)(base);
            bf16x8 hi = *(const bf16x8*)(base + 16);
            wfrag[ni] = __builtin_shufflevector(lo, hi,
                0,1,2,3,4,5,6,7,8,9,10,11,12,13,14,15);
        }

#pragma unroll
        for (int mi = 0; mi < 2; ++mi)
#pragma unroll
            for (int ni = 0; ni < 4; ++ni)
                acc[mi][ni] = __builtin_amdgcn_wmma_f32_16x16x32_bf16(
                    false, afrag[mi], false, wfrag[ni],
                    (short)0, acc[mi][ni], false, false);

        if (more) wait_async0();   // next tile's async W copy done before barrier
        __syncthreads();
        buf ^= 1;
    }

    // ---- epilogue: C/D layout -> M = vgpr + 8*(lane>=16), N = lane&15
    const int nlo = lane & 15;
    const int mhi = (lane >> 4) * 8;
#pragma unroll
    for (int mi = 0; mi < 2; ++mi) {
#pragma unroll
        for (int ni = 0; ni < 4; ++ni) {
            const int gn = bn + wn * 64 + ni * 16 + nlo;
            const float bv = bias[gn];
#pragma unroll
            for (int r = 0; r < 8; ++r) {
                const int gm = bm + wm * 32 + mi * 16 + mhi + r;
                float v = acc[mi][ni][r] + bv;
                if (act == 1) v = gelu_exact(v);
                C[(size_t)gm * ldc + gn] = v;
            }
        }
    }
}

// ---------------------------------------------------------------------------
// RoPE: hr = rope(h); one block per token row (D=512 -> 256 pairs)
// ---------------------------------------------------------------------------
__global__ __launch_bounds__(256) void rope_kernel(
    const float* __restrict__ h, float* __restrict__ hr)
{
    const int m    = blockIdx.x;          // token row 0..2047
    const int col0 = threadIdx.x * 2;     // even element index within D
    const int p    = (col0 & 63) >> 1;    // rotary pair index within head (0..31)
    const int s    = m & 31;              // sequence position
    const float theta = __powf(10000.0f, -(float)(2 * p) * (1.0f / 64.0f));
    float sn, cs;
    __sincosf((float)s * theta, &sn, &cs);
    const size_t base = (size_t)m * 512;
    const float x0 = h[base + col0];
    const float x1 = h[base + col0 + 1];
    hr[base + col0]     = x0 * cs - x1 * sn;
    hr[base + col0 + 1] = x1 * cs + x0 * sn;
}

// ---------------------------------------------------------------------------
// Causal attention per (b, head): S=32, HD=64. qkv rows are [q(512)|k(512)|v(512)].
// ---------------------------------------------------------------------------
__global__ __launch_bounds__(256) void attention_kernel(
    const float* __restrict__ qkv, float* __restrict__ ctx)
{
    const int b  = blockIdx.x >> 3;
    const int hh = blockIdx.x & 7;
    const int tid = threadIdx.x;

    __shared__ float qs[32][65], ks[32][65], vs[32][65];
    __shared__ float sc[32][33];

    for (int i = tid; i < 2048; i += 256) {
        const int s = i >> 6, d = i & 63;
        const size_t row = (size_t)(b * 32 + s) * 1536;
        qs[s][d] = qkv[row +        hh * 64 + d];
        ks[s][d] = qkv[row +  512 + hh * 64 + d];
        vs[s][d] = qkv[row + 1024 + hh * 64 + d];
    }
    __syncthreads();

    const float scale = 0.125f;  // 1/sqrt(64)
    for (int i = tid; i < 1024; i += 256) {
        const int qi = i >> 5, kj = i & 31;
        float a = 0.0f;
        if (kj <= qi) {
#pragma unroll 8
            for (int d = 0; d < 64; ++d) a += qs[qi][d] * ks[kj][d];
            a *= scale;
        } else {
            a = -3.0e38f;
        }
        sc[qi][kj] = a;
    }
    __syncthreads();

    if (tid < 32) {  // row softmax (causal)
        const int row = tid;
        float mx = -3.0e38f;
        for (int j = 0; j <= row; ++j) mx = fmaxf(mx, sc[row][j]);
        float sum = 0.0f;
        for (int j = 0; j < 32; ++j) {
            const float e = (j <= row) ? __expf(sc[row][j] - mx) : 0.0f;
            sc[row][j] = e;
            sum += e;
        }
        const float inv = 1.0f / sum;
        for (int j = 0; j < 32; ++j) sc[row][j] *= inv;
    }
    __syncthreads();

    for (int i = tid; i < 2048; i += 256) {
        const int s = i >> 6, d = i & 63;
        float a = 0.0f;
#pragma unroll 8
        for (int j = 0; j < 32; ++j) a += sc[s][j] * vs[j][d];
        ctx[(size_t)(b * 32 + s) * 512 + hh * 64 + d] = a;
    }
}

// ---------------------------------------------------------------------------
// out[m] = LayerNorm(x[m] + r[m]) * w + b   (D=512, one block per row; safe
// in-place on x since each block only touches its own row)
// ---------------------------------------------------------------------------
__global__ __launch_bounds__(256) void add_ln_kernel(
    const float* x, const float* r,
    const float* __restrict__ w, const float* __restrict__ bb, float* out)
{
    const int m   = blockIdx.x;
    const int tid = threadIdx.x;
    __shared__ float red[256];
    const size_t base = (size_t)m * 512;

    const float v0 = x[base + tid]       + r[base + tid];
    const float v1 = x[base + 256 + tid] + r[base + 256 + tid];

    red[tid] = v0 + v1;
    __syncthreads();
    for (int st = 128; st > 0; st >>= 1) {
        if (tid < st) red[tid] += red[tid + st];
        __syncthreads();
    }
    const float mu = red[0] * (1.0f / 512.0f);
    __syncthreads();

    const float d0 = v0 - mu, d1 = v1 - mu;
    red[tid] = d0 * d0 + d1 * d1;
    __syncthreads();
    for (int st = 128; st > 0; st >>= 1) {
        if (tid < st) red[tid] += red[tid + st];
        __syncthreads();
    }
    const float inv = rsqrtf(red[0] * (1.0f / 512.0f) + 1e-5f);

    out[base + tid]       = d0 * inv * w[tid]       + bb[tid];
    out[base + 256 + tid] = d1 * inv * w[256 + tid] + bb[256 + tid];
}

// ---------------------------------------------------------------------------
// Orchestration
// ---------------------------------------------------------------------------
static inline void convert_w(const float* src, bf16_t* dst, int n, hipStream_t s) {
    const int n4 = n >> 2;
    f32_to_bf16_kernel<<<dim3((n4 + 255) / 256), dim3(256), 0, s>>>(src, dst, n4);
}

extern "C" void kernel_launch(void* const* d_in, const int* in_sizes, int n_in,
                              void* d_out, int out_size, void* d_ws, size_t ws_size,
                              hipStream_t stream) {
    const float* x    = (const float*)d_in[0];
    const float* ipw0 = (const float*)d_in[1];   // inp_w (512, 9216)
    const float* ipb0 = (const float*)d_in[2];
    const float* ipw  = (const float*)d_in[3];   // in_proj_w (L, 1536, 512)
    const float* ipb  = (const float*)d_in[4];
    const float* ow   = (const float*)d_in[5];   // out_w (L, 512, 512)
    const float* ob   = (const float*)d_in[6];
    const float* ln1w = (const float*)d_in[7];
    const float* ln1b = (const float*)d_in[8];
    const float* ln2w = (const float*)d_in[9];
    const float* ln2b = (const float*)d_in[10];
    const float* f1w  = (const float*)d_in[11];  // (L, 2048, 512)
    const float* f1b  = (const float*)d_in[12];
    const float* f2w  = (const float*)d_in[13];  // (L, 512, 2048)
    const float* f2b  = (const float*)d_in[14];
    const float* opw  = (const float*)d_in[15];  // (9216, 512)
    const float* opb  = (const float*)d_in[16];
    float* out = (float*)d_out;

    const int M = 2048, D = 512, IN = 9216, FF = 2048, L = 6;

    // workspace layout (~56 MB)
    float* h   = (float*)d_ws;
    float* hr  = h   + (size_t)M * D;
    float* qkv = hr  + (size_t)M * D;
    float* ctx = qkv + (size_t)M * 3 * D;
    float* tmp = ctx + (size_t)M * D;
    float* ffm = tmp + (size_t)M * D;
    bf16_t* wbuf = (bf16_t*)(ffm + (size_t)M * FF);   // up to 512*9216 bf16

    dim3 blk(256);

    // h = x @ inp_w^T + inp_b        (2048 x 9216 x 512)
    convert_w(ipw0, wbuf, D * IN, stream);
    gemm_bias_act_kernel<<<dim3(D / BN, M / BM), blk, 0, stream>>>(
        x, wbuf, ipb0, h, M, D, IN, D, 0);

    for (int i = 0; i < L; ++i) {
        const float* bqk = ipb + (size_t)i * 3 * D;
        const float* bv  = bqk + 2 * D;

        // convert this layer's fused qkv weights once (reused 16x by M-tiles)
        convert_w(ipw + (size_t)i * 3 * D * D, wbuf, 3 * D * D, stream);

        rope_kernel<<<dim3(M), blk, 0, stream>>>(h, hr);
        // q,k from rotated h (rows 0..1023 of in_proj_w)
        gemm_bias_act_kernel<<<dim3(2 * D / BN, M / BM), blk, 0, stream>>>(
            hr, wbuf, bqk, qkv, M, 2 * D, D, 3 * D, 0);
        // v from un-rotated h (rows 1024..1535)
        gemm_bias_act_kernel<<<dim3(D / BN, M / BM), blk, 0, stream>>>(
            h, wbuf + (size_t)2 * D * D, bv, qkv + 2 * D, M, D, D, 3 * D, 0);

        attention_kernel<<<dim3(512), blk, 0, stream>>>(qkv, ctx);

        convert_w(ow + (size_t)i * D * D, wbuf, D * D, stream);
        gemm_bias_act_kernel<<<dim3(D / BN, M / BM), blk, 0, stream>>>(
            ctx, wbuf, ob + (size_t)i * D, tmp, M, D, D, D, 0);
        add_ln_kernel<<<dim3(M), blk, 0, stream>>>(
            h, tmp, ln1w + (size_t)i * D, ln1b + (size_t)i * D, h);

        convert_w(f1w + (size_t)i * FF * D, wbuf, FF * D, stream);
        gemm_bias_act_kernel<<<dim3(FF / BN, M / BM), blk, 0, stream>>>(
            h, wbuf, f1b + (size_t)i * FF, ffm, M, FF, D, FF, 1);

        convert_w(f2w + (size_t)i * D * FF, wbuf, D * FF, stream);
        gemm_bias_act_kernel<<<dim3(D / BN, M / BM), blk, 0, stream>>>(
            ffm, wbuf, f2b + (size_t)i * D, tmp, M, D, FF, D, 0);
        add_ln_kernel<<<dim3(M), blk, 0, stream>>>(
            h, tmp, ln2w + (size_t)i * D, ln2b + (size_t)i * D, h);
    }

    // out = h @ outp_w^T + outp_b    (2048 x 512 x 9216)
    convert_w(opw, wbuf, IN * D, stream);
    gemm_bias_act_kernel<<<dim3(IN / BN, M / BM), blk, 0, stream>>>(
        h, wbuf, opb, out, M, IN, D, IN, 0);

    (void)in_sizes; (void)n_in; (void)out_size; (void)ws_size;
}
